// DGCNNEncoder_73383811219651
// MI455X (gfx1250) — matrix operand order repaired and checked by reading, not compiled, not run on hardware
//
#include <hip/hip_runtime.h>

typedef __attribute__((ext_vector_type(16))) _Float16 v16h;
typedef __attribute__((ext_vector_type(8)))  _Float16 v8h;
typedef __attribute__((ext_vector_type(8)))  float    v8f;

#define LRELU_SLOPE 0.2f
#define BNSCALE 0.99999500003749972f   /* 1/sqrt(1+1e-5) */
#define KEY_NEG_INF ((int)0x807fffffu) /* monotonic int key of -inf */

static __device__ __forceinline__ float lrelu(float x) { return x > 0.f ? x : LRELU_SLOPE * x; }
// order-preserving float<->int key (all finite values, both signs)
static __device__ __forceinline__ int   fkey(float f)  { int i = __float_as_int(f); return i >= 0 ? i : (i ^ 0x7fffffff); }
static __device__ __forceinline__ float kfloat(int k)  { return __int_as_float(k >= 0 ? k : (k ^ 0x7fffffff)); }

// load two aligned 16-byte half8 runs into one WMMA fragment register set
static __device__ __forceinline__ v16h ld_pair(const _Float16* p0, const _Float16* p1) {
  union { v16h v; v8h h[2]; } u;
  u.h[0] = *(const v8h*)p0;
  u.h[1] = *(const v8h*)p1;
  return u.v;
}

// ---------------------------------------------------------------- points (B,N,3) -> xh0[b][n][32] f16 (padded) + norms
__global__ void k_pts(const float* __restrict__ pts, _Float16* __restrict__ xh0,
                      float* __restrict__ norms, int N) {
  int b = blockIdx.y;
  int n = blockIdx.x * blockDim.x + threadIdx.x;
  if (n >= N) return;
  float x = pts[((long)b * N + n) * 3 + 0];
  float y = pts[((long)b * N + n) * 3 + 1];
  float z = pts[((long)b * N + n) * 3 + 2];
  _Float16* r = xh0 + ((long)b * N + n) * 32;
  r[0] = (_Float16)x; r[1] = (_Float16)y; r[2] = (_Float16)z;
  for (int c = 3; c < 32; ++c) r[c] = (_Float16)0.f;
  norms[b * N + n] = x * x + y * y + z * z;
}

// ---------------------------------------------------------------- weight f32 -> zero-padded f16 (row stride rspad)
__global__ void k_wpad(const float* __restrict__ W, _Float16* __restrict__ Wh,
                       int rows, int cols, int rspad) {
  int i = blockIdx.x * blockDim.x + threadIdx.x;
  if (i >= rows * rspad) return;
  int r = i / rspad, c = i % rspad;
  Wh[i] = (c < cols) ? (_Float16)W[(long)r * cols + c] : (_Float16)0.f;
}

// ---------------------------------------------------------------- per-point squared norms from f16 rows
__global__ void k_norms(const _Float16* __restrict__ xh, long rs, int C,
                        float* __restrict__ norms, int N) {
  int b = blockIdx.y;
  int n = blockIdx.x * blockDim.x + threadIdx.x;
  if (n >= N) return;
  const _Float16* p = xh + ((long)b * N + n) * rs;
  float s = 0.f;
  for (int c = 0; c < C; ++c) { float v = (float)p[c]; s += v * v; }
  norms[b * N + n] = s;
}

// ---------------------------------------------------------------- KNN: WMMA Gram + fused top-20
// block = 512 thr (16 waves) handles (b, 16-row strip). xh rows are fragment-native:
// A frag = two b128 loads, B frag = two b128 loads, then v_wmma. pdist strip in LDS.
__global__ __launch_bounds__(512) void k_knn(const _Float16* __restrict__ xh, long rs, int C,
                                             const float* __restrict__ norms,
                                             int* __restrict__ knn_out, int N) {
  extern __shared__ char smem[];
  const int NS = N + 16;                         // padded row stride: bank-conflict-free
  float* pd = (float*)smem;                      // 16*NS f32
  float* rn = (float*)(smem + (size_t)16 * NS * 4);

  const int b = blockIdx.y, n0 = blockIdx.x * 16;
  const int tid = threadIdx.x, wave = tid >> 5, lane = tid & 31;
  if (tid < 16) rn[tid] = norms[b * N + n0 + tid];
  __syncthreads();

  const int base8 = (lane & 16) ? 8 : 0;         // ISA f16-A lane offset
  const int bb    = (lane & 16) ? 16 : 0;        // unused: B is loaded with base8, like A
  const _Float16* arow = xh + ((long)b * N + n0 + (lane & 15)) * rs;

  for (int t = wave; t < (N >> 4); t += 16) {    // uniform per wave -> EXEC all-ones at WMMA
    const int m0 = t << 4;
    const _Float16* brow = xh + ((long)b * N + m0 + (lane & 15)) * rs;
    v8f acc = {};
    for (int kc = 0; kc < C; kc += 32) {
      v16h A  = ld_pair(arow + kc + base8, arow + kc + 16 + base8);
      v16h Bm = ld_pair(brow + kc + base8, brow + kc + 16 + base8);
      acc = __builtin_amdgcn_wmma_f32_16x16x32_f16(false, A, false, Bm, (short)0, acc,
                                                   false, false);
    }
    const int col = lane & 15;
    const float cn = norms[b * N + m0 + col];
#pragma unroll
    for (int r = 0; r < 8; ++r) {                // C/D: row = r + (lane>=16 ? 8 : 0)
      int row = r + ((lane & 16) ? 8 : 0);
      pd[row * NS + m0 + col] = 2.f * acc[r] - rn[row] - cn;  // negative sq-dist
    }
  }
  __syncthreads();

  // fused top-20 (largest = nearest, incl. self). wave w owns row w. deterministic ties.
  {
    const int row = wave;
    const int PER = N >> 5;
    for (int t = 0; t < 20; ++t) {
      float bv = -3.4e38f;
      int   bc = N;
      for (int i = 0; i < PER; ++i) {
        int   c = lane + (i << 5);
        float v = pd[row * NS + c];
        if (v > bv || (v == bv && c < bc)) { bv = v; bc = c; }
      }
#pragma unroll
      for (int m = 16; m >= 1; m >>= 1) {
        float ov = __shfl_xor(bv, m);
        int   oc = __shfl_xor(bc, m);
        if (ov > bv || (ov == bv && oc < bc)) { bv = ov; bc = oc; }
      }
      if ((bc & 31) == lane) pd[row * NS + bc] = -3.4e38f;
      if (lane == 0) knn_out[((long)b * N + n0 + row) * 20 + t] = bc;
    }
  }
}

// ---------------------------------------------------------------- EdgeConv: gather + WMMA + BN/LReLU + max-over-K
// block = 512 thr handles (b, 8 points) = 160 edge columns = 10 WMMA col-tiles.
// Wh is f16, zero-padded, row stride C2P. Output written f16 into fh (row stride 512).
template <int C2P>
__global__ __launch_bounds__(512) void k_edgeconv(const _Float16* __restrict__ xh, long rs, int C,
                                                  const int* __restrict__ knn,
                                                  const _Float16* __restrict__ Wh,
                                                  const float* __restrict__ gam,
                                                  const float* __restrict__ bet,
                                                  _Float16* __restrict__ yout, /* fh + choff */
                                                  int Cout, int N) {
  extern __shared__ char smem[];
  const int ER = C2P + 8;
  _Float16* edg  = (_Float16*)smem;                                                // 160*ER
  int*      ymax = (int*)(smem + ((((size_t)160 * ER * 2) + 255) & ~(size_t)255)); // Cout*8

  const int b = blockIdx.y, n0 = blockIdx.x * 8;
  const int tid = threadIdx.x, wave = tid >> 5, lane = tid & 31;
  const int C2 = 2 * C;

  for (int e = tid; e < Cout * 8; e += 512) ymax[e] = KEY_NEG_INF;

  // stage edge features [col][cc]: cc<C center, C..2C-1 (neighbor-center), rest zero
  for (int col = wave; col < 160; col += 16) {
    const int nl = col / 20, ki = col % 20;
    const long rown = ((long)b * N + n0 + nl) * rs;
    const int  j    = knn[((long)b * N + n0 + nl) * 20 + ki];
    const long rowj = ((long)b * N + j) * rs;
    for (int c = lane; c < C2P; c += 32) {
      _Float16 v;
      if (c < C)       v = xh[rown + c];
      else if (c < C2) v = (_Float16)(xh[rowj + (c - C)] - xh[rown + (c - C)]);
      else             v = (_Float16)0.f;
      edg[col * ER + c] = v;
    }
  }
  __syncthreads();

  const int nM = Cout >> 4;
  const int njobs = nM * 10;
  const int base8 = (lane & 16) ? 8 : 0;
  const int bb    = (lane & 16) ? 16 : 0;
  for (int job = wave; job < njobs; job += 16) {
    const int mt = job / 10, ct = job % 10;
    const _Float16* wrow = Wh + (long)(mt * 16 + (lane & 15)) * C2P;
    const _Float16* ecol = edg + (ct * 16 + (lane & 15)) * ER;
    v8f acc = {};
#pragma unroll
    for (int kc = 0; kc < C2P; kc += 32) {
      v16h A  = ld_pair(wrow + kc + base8, wrow + kc + 16 + base8);
      v16h Bm = ld_pair(ecol + kc + base8, ecol + kc + 16 + base8);
      acc = __builtin_amdgcn_wmma_f32_16x16x32_f16(false, A, false, Bm, (short)0, acc,
                                                   false, false);
    }
    const int col = ct * 16 + (lane & 15);
    const int nl = col / 20;
#pragma unroll
    for (int r = 0; r < 8; ++r) {
      int o = mt * 16 + r + ((lane & 16) ? 8 : 0);
      float h = acc[r] * (gam[o] * BNSCALE) + bet[o];
      h = lrelu(h);
      atomicMax(&ymax[o * 8 + nl], fkey(h));   // int max: order-independent => deterministic
    }
  }
  __syncthreads();
  // write f16 rows of fh (coalesced along channels)
  for (int e = tid; e < Cout * 8; e += 512) {
    int nl = e / Cout, o = e % Cout;
    yout[((long)b * N + n0 + nl) * 512 + o] = (_Float16)kfloat(ymax[o * 8 + nl]);
  }
}

// ---------------------------------------------------------------- init global max keys
__global__ void k_init(int* __restrict__ gmax, int n) {
  int i = blockIdx.x * blockDim.x + threadIdx.x;
  if (i < n) gmax[i] = KEY_NEG_INF;
}

// ---------------------------------------------------------------- conv5 (1024x512) + BN/LReLU + max/mean pooling
// block = 512 thr: (b, 16-o tile, 256-n chunk). Operands fragment-native in memory:
// pure b128 loads + v_wmma. Max via int-key atomics; mean via fixed-order partials.
__global__ __launch_bounds__(512) void k_conv5(const _Float16* __restrict__ fh,
                                               const _Float16* __restrict__ w5h,
                                               const float* __restrict__ g5,
                                               const float* __restrict__ b5,
                                               int* __restrict__ gmax,
                                               float* __restrict__ psum, int N) {
  __shared__ float wsum[16][16];   // [o_local][wave]
  const int b = blockIdx.z, ot = blockIdx.x, nc = blockIdx.y;
  const int tid = threadIdx.x, wave = tid >> 5, lane = tid & 31;
  const int base8 = (lane & 16) ? 8 : 0;
  const int bb    = (lane & 16) ? 16 : 0;
  const _Float16* wrow = w5h + (long)(ot * 16 + (lane & 15)) * 512;
  const _Float16* frow = fh + ((long)b * N + nc * 256 + wave * 16 + (lane & 15)) * 512;
  v8f acc = {};
#pragma unroll 4
  for (int kc = 0; kc < 512; kc += 32) {
    v16h A  = ld_pair(wrow + kc + base8, wrow + kc + 16 + base8);
    v16h Bm = ld_pair(frow + kc + base8, frow + kc + 16 + base8);
    acc = __builtin_amdgcn_wmma_f32_16x16x32_f16(false, A, false, Bm, (short)0, acc,
                                                 false, false);
  }
#pragma unroll
  for (int r = 0; r < 8; ++r) {
    int o = ot * 16 + r + ((lane & 16) ? 8 : 0);
    float h = acc[r] * (g5[o] * BNSCALE) + b5[o];
    h = lrelu(h);
    float mx = h, sm = h;
#pragma unroll
    for (int m = 8; m >= 1; m >>= 1) {       // stays within 16-lane half
      float om = __shfl_xor(mx, m);
      mx = om > mx ? om : mx;
      sm += __shfl_xor(sm, m);
    }
    if ((lane & 15) == r) {
      atomicMax(&gmax[b * 1024 + o], fkey(mx));
      wsum[r + ((lane & 16) ? 8 : 0)][wave] = sm;
    }
  }
  __syncthreads();
  if (tid < 16) {                            // fixed-order sum over 16 waves
    float s = 0.f;
    for (int w = 0; w < 16; ++w) s += wsum[tid][w];
    psum[((long)b * 1024 + ot * 16 + tid) * 8 + nc] = s;
  }
}

// ---------------------------------------------------------------- pooling finalize -> f (B,2048)
__global__ void k_pool_fin(const int* __restrict__ gmax, const float* __restrict__ psum,
                           float* __restrict__ f, int N) {
  int b = blockIdx.y;
  int o = blockIdx.x * blockDim.x + threadIdx.x;
  if (o >= 1024) return;
  f[b * 2048 + o] = kfloat(gmax[b * 1024 + o]);
  float s = 0.f;
  for (int nc = 0; nc < 8; ++nc) s += psum[((long)b * 1024 + o) * 8 + nc];  // fixed order
  f[b * 2048 + 1024 + o] = s * (1.0f / N);
}

// ---------------------------------------------------------------- FC layers (tiny): wave per output, fp32
__global__ void k_fc1(const float* __restrict__ f, const float* __restrict__ fw,
                      const float* __restrict__ fb, const float* __restrict__ g,
                      const float* __restrict__ be, float* __restrict__ out) {
  int gw = (blockIdx.x * blockDim.x + threadIdx.x) >> 5;
  int lane = threadIdx.x & 31;
  int b = gw >> 9, o = gw & 511;
  float s = 0.f;
  for (int c = lane; c < 2048; c += 32) s += f[b * 2048 + c] * fw[(long)o * 2048 + c];
#pragma unroll
  for (int m = 16; m >= 1; m >>= 1) s += __shfl_xor(s, m);
  if (lane == 0) {
    float h = (s + fb[o]) * (g[o] * BNSCALE) + be[o];
    out[b * 512 + o] = lrelu(h);
  }
}

__global__ void k_fc2(const float* __restrict__ f, const float* __restrict__ fw,
                      const float* __restrict__ fb, const float* __restrict__ g,
                      const float* __restrict__ be, float* __restrict__ out) {
  int gw = (blockIdx.x * blockDim.x + threadIdx.x) >> 5;
  int lane = threadIdx.x & 31;
  int b = gw >> 8, o = gw & 255;
  float s = 0.f;
  for (int c = lane; c < 512; c += 32) s += f[b * 512 + c] * fw[(long)o * 512 + c];
#pragma unroll
  for (int m = 16; m >= 1; m >>= 1) s += __shfl_xor(s, m);
  if (lane == 0) out[b * 256 + o] = (s + fb[o]) * (g[o] * BNSCALE) + be[o];
}

// ================================================================ host launcher
extern "C" void kernel_launch(void* const* d_in, const int* in_sizes, int n_in,
                              void* d_out, int out_size, void* d_ws, size_t ws_size,
                              hipStream_t stream) {
  (void)in_sizes; (void)n_in; (void)out_size; (void)ws_size;
  const int B = 4, N = 2048;

  const float* pts = (const float*)d_in[0];
  const float *w1 = (const float*)d_in[1],  *g1 = (const float*)d_in[2],  *b1 = (const float*)d_in[3];
  const float *w2 = (const float*)d_in[4],  *g2 = (const float*)d_in[5],  *b2 = (const float*)d_in[6];
  const float *w3 = (const float*)d_in[7],  *g3 = (const float*)d_in[8],  *b3 = (const float*)d_in[9];
  const float *w4 = (const float*)d_in[10], *g4 = (const float*)d_in[11], *b4 = (const float*)d_in[12];
  const float *w5 = (const float*)d_in[13], *g5 = (const float*)d_in[14], *b5 = (const float*)d_in[15];
  const float *fw1 = (const float*)d_in[16], *fb1 = (const float*)d_in[17];
  const float *g6 = (const float*)d_in[18], *b6 = (const float*)d_in[19];
  const float *fw2 = (const float*)d_in[20], *fb2 = (const float*)d_in[21];
  const float *g7 = (const float*)d_in[22], *b7 = (const float*)d_in[23];

  char* ws = (char*)d_ws;
  size_t off = 0;
  auto carve = [&](size_t bytes) -> char* {
    char* p = ws + off;
    off = (off + bytes + 255) & ~(size_t)255;
    return p;
  };
  _Float16* xh0  = (_Float16*)carve((size_t)B * N * 32 * 2);
  _Float16* fh   = (_Float16*)carve((size_t)B * N * 512 * 2);
  float*    norms= (float*)carve((size_t)B * N * 4);
  int*      idx  = (int*)carve((size_t)B * N * 20 * 4);
  int*      gmax = (int*)carve((size_t)B * 1024 * 4);
  float*    psum = (float*)carve((size_t)B * 1024 * 8 * 4);
  float*    fvec = (float*)carve((size_t)B * 2048 * 4);
  float*    ff1  = (float*)carve((size_t)B * 512 * 4);
  _Float16* wh1  = (_Float16*)carve((size_t)64 * 32 * 2);
  _Float16* wh2  = (_Float16*)carve((size_t)64 * 128 * 2);
  _Float16* wh3  = (_Float16*)carve((size_t)128 * 128 * 2);
  _Float16* wh4  = (_Float16*)carve((size_t)256 * 256 * 2);
  _Float16* wh5  = (_Float16*)carve((size_t)1024 * 512 * 2);

  // points -> padded f16 + norms; weights -> padded f16
  k_pts<<<dim3(N / 256, B), 256, 0, stream>>>(pts, xh0, norms, N);
  k_wpad<<<(64 * 32 + 255) / 256, 256, 0, stream>>>(w1, wh1, 64, 6, 32);
  k_wpad<<<(64 * 128 + 255) / 256, 256, 0, stream>>>(w2, wh2, 64, 128, 128);
  k_wpad<<<(128 * 128 + 255) / 256, 256, 0, stream>>>(w3, wh3, 128, 128, 128);
  k_wpad<<<(256 * 256 + 255) / 256, 256, 0, stream>>>(w4, wh4, 256, 256, 256);
  k_wpad<<<(1024 * 512 + 255) / 256, 256, 0, stream>>>(w5, wh5, 1024, 512, 512);

  const size_t knn_lds = (size_t)16 * (N + 16) * 4 + 64;
  auto edge_lds = [&](int C2P, int Cout) {
    return ((((size_t)160 * (C2P + 8) * 2) + 255) & ~(size_t)255) + (size_t)Cout * 8 * 4;
  };

  // ---- EdgeConv 1: xh0 (C=3, padded 32) -> fh channels [0..64)
  k_knn<<<dim3(N / 16, B), 512, knn_lds, stream>>>(xh0, 32, 32, norms, idx, N);
  k_edgeconv<32><<<dim3(N / 8, B), 512, edge_lds(32, 64), stream>>>(
      xh0, 32, 3, idx, wh1, g1, b1, fh, 64, N);

  // ---- EdgeConv 2: fh[0..64) -> fh[64..128)
  k_norms<<<dim3(N / 256, B), 256, 0, stream>>>(fh, 512, 64, norms, N);
  k_knn<<<dim3(N / 16, B), 512, knn_lds, stream>>>(fh, 512, 64, norms, idx, N);
  k_edgeconv<128><<<dim3(N / 8, B), 512, edge_lds(128, 64), stream>>>(
      fh, 512, 64, idx, wh2, g2, b2, fh + 64, 64, N);

  // ---- EdgeConv 3: fh[64..128) -> fh[128..256)
  k_norms<<<dim3(N / 256, B), 256, 0, stream>>>(fh + 64, 512, 64, norms, N);
  k_knn<<<dim3(N / 16, B), 512, knn_lds, stream>>>(fh + 64, 512, 64, norms, idx, N);
  k_edgeconv<128><<<dim3(N / 8, B), 512, edge_lds(128, 128), stream>>>(
      fh + 64, 512, 64, idx, wh3, g3, b3, fh + 128, 128, N);

  // ---- EdgeConv 4: fh[128..256) -> fh[256..512)
  k_norms<<<dim3(N / 256, B), 256, 0, stream>>>(fh + 128, 512, 128, norms, N);
  k_knn<<<dim3(N / 16, B), 512, knn_lds, stream>>>(fh + 128, 512, 128, norms, idx, N);
  k_edgeconv<256><<<dim3(N / 8, B), 512, edge_lds(256, 256), stream>>>(
      fh + 128, 512, 128, idx, wh4, g4, b4, fh + 256, 256, N);

  // ---- conv5 + pooling
  k_init<<<(B * 1024 + 255) / 256, 256, 0, stream>>>(gmax, B * 1024);
  k_conv5<<<dim3(64, 8, B), 512, 0, stream>>>(fh, wh5, g5, b5, gmax, psum, N);
  k_pool_fin<<<dim3(4, B), 256, 0, stream>>>(gmax, psum, fvec, N);

  // ---- FC head
  k_fc1<<<(B * 512 * 32) / 256, 256, 0, stream>>>(fvec, fw1, fb1, g6, b6, ff1);
  k_fc2<<<(B * 256 * 32) / 256, 256, 0, stream>>>(ff1, fw2, fb2, g7, b7, (float*)d_out);
}
